// VectorQuantizerEMA_21019569946794
// MI455X (gfx1250) — matrix operand order repaired and checked by reading, not compiled, not run on hardware
//
#include <hip/hip_runtime.h>
#include <hip/hip_fp16.h>
#include <math.h>

typedef __attribute__((ext_vector_type(16))) _Float16 v16h;
typedef __attribute__((ext_vector_type(8)))  _Float16 v8h;
typedef __attribute__((ext_vector_type(8)))  float    v8f;

#define N_PTS    131072      // B*H*W = 32*64*64
#define DIMV     64          // D (= C)
#define K_CODES  1024
#define HWSZ     4096        // H*W
#define CHW      262144      // C*H*W
#define OUT_ELEMS 8388608    // B*C*H*W
#define ROW_H    72          // padded LDS row pitch in halves (144 B)

// ---------------------------------------------------------------------------
// Kernel 1: zero encodings region, build f16 codebook + ||e||^2, zero counters
// ---------------------------------------------------------------------------
__global__ void vq_prep_kernel(const float* __restrict__ emb,
                               float2* __restrict__ enc2, size_t npairs,
                               _Float16* __restrict__ emb_h,
                               float* __restrict__ e2,
                               unsigned int* __restrict__ hist,
                               float* __restrict__ sqsum) {
    size_t gid    = (size_t)blockIdx.x * blockDim.x + threadIdx.x;
    size_t stride = (size_t)gridDim.x * blockDim.x;
    float2 z2; z2.x = 0.0f; z2.y = 0.0f;
    for (size_t i = gid; i < npairs; i += stride) enc2[i] = z2;
    if (gid < K_CODES) {
        float s = 0.0f;
        #pragma unroll 8
        for (int d = 0; d < DIMV; ++d) {
            float v = emb[gid * DIMV + d];
            s += v * v;
            emb_h[gid * DIMV + d] = (_Float16)v;
        }
        e2[gid]   = s;
        hist[gid] = 0u;
        if (gid == 0) sqsum[0] = 0.0f;
    }
}

// ---------------------------------------------------------------------------
// Kernel 2: distances via WMMA f16 (f32 accum), argmin, scatter one-hot,
//           gather quantized -> out [B,C,H,W], loss partials, histogram.
// One wave (32 lanes) handles a tile of 16 consecutive z rows vs all 1024
// codes. Whole f16 codebook staged in LDS with 144B row pitch.
// ---------------------------------------------------------------------------
__global__ void vq_main_kernel(const float* __restrict__ z,
                               const float* __restrict__ embf,
                               const _Float16* __restrict__ emb_h,
                               const float* __restrict__ e2g,
                               unsigned int* __restrict__ hist,
                               float* __restrict__ sqsum,
                               float* __restrict__ outp,
                               float* __restrict__ enc) {
    extern __shared__ char smem[];
    _Float16* embl = (_Float16*)smem;                          // 1024*72 halves = 147456 B
    float*    e2l  = (float*)(smem + K_CODES * ROW_H * 2);     // 4096 B
    int*      widx = (int*)(smem + K_CODES * ROW_H * 2 + 4096);// 8*16*4 = 512 B

    const int tid  = threadIdx.x;
    const int w    = tid >> 5;
    const int lane = tid & 31;

    // --- stage f16 codebook into LDS (padded rows), 16B chunks ---
    for (int i = tid; i < K_CODES * 8; i += 256) {
        int row = i >> 3, cp = i & 7;
        *(float4*)(embl + row * ROW_H + cp * 8) =
            *(const float4*)(emb_h + row * DIMV + cp * 8);
    }
    for (int i = tid; i < K_CODES; i += 256) e2l[i] = e2g[i];
    __syncthreads();

    // --- tile setup ---
    const int tile = blockIdx.x * 8 + w;        // 8192 tiles total
    const int n0   = tile * 16;
    const int bimg = n0 >> 12;                  // n0 / HWSZ
    const int hw0  = n0 & (HWSZ - 1);
    const int m    = lane & 15;                 // M row (A) / N col (B,C,D)
    const int moff = (lane < 16) ? 0 : 8;       // A-frag K group offset
    const int koff = (lane < 16) ? 0 : 8;       // B-frag K group offset
    const size_t zrowbase = (size_t)bimg * CHW + hw0;

    // --- load A fragments (16 rows x 64 dims, f32 -> f16), reused 64x ---
    v16h a0, a1;
    #pragma unroll
    for (int v = 0; v < 8; ++v) {
        int kb = 2 * (v & 3) + 16 * (v >> 2) + moff;
        a0[2 * v]     = (_Float16)z[zrowbase + (size_t)(kb + 0) * HWSZ + m];
        a0[2 * v + 1] = (_Float16)z[zrowbase + (size_t)(kb + 1) * HWSZ + m];
        a1[2 * v]     = (_Float16)z[zrowbase + (size_t)(kb + 32) * HWSZ + m];
        a1[2 * v + 1] = (_Float16)z[zrowbase + (size_t)(kb + 33) * HWSZ + m];
    }

    float minv[8];
    int   mini[8];
    #pragma unroll
    for (int v = 0; v < 8; ++v) { minv[v] = 3.4e38f; mini[v] = 0; }

    // --- sweep all 64 code tiles ---
    for (int c0 = 0; c0 < K_CODES; c0 += 16) {
        const _Float16* rb = embl + (c0 + m) * ROW_H + koff;
        v8h lo0 = *(const v8h*)(rb);
        v8h hi0 = *(const v8h*)(rb + 16);
        v8h lo1 = *(const v8h*)(rb + 32);
        v8h hi1 = *(const v8h*)(rb + 48);
        v16h b0 = __builtin_shufflevector(lo0, hi0, 0,1,2,3,4,5,6,7,8,9,10,11,12,13,14,15);
        v16h b1 = __builtin_shufflevector(lo1, hi1, 0,1,2,3,4,5,6,7,8,9,10,11,12,13,14,15);

        v8f acc = {};
        acc = __builtin_amdgcn_wmma_f32_16x16x32_f16(false, a0, false, b0,
                                                     (short)0, acc, false, false);
        acc = __builtin_amdgcn_wmma_f32_16x16x32_f16(false, a1, false, b1,
                                                     (short)0, acc, false, false);

        float e2v = e2l[c0 + m];
        int   cid = c0 + m;
        #pragma unroll
        for (int v = 0; v < 8; ++v) {
            float s = fmaf(-2.0f, acc[v], e2v);
            if (s < minv[v]) { minv[v] = s; mini[v] = cid; }
        }
    }

    // --- argmin across the 16 lanes of each half (xor 1,2,4,8) ---
    #pragma unroll
    for (int off = 1; off <= 8; off <<= 1) {
        #pragma unroll
        for (int v = 0; v < 8; ++v) {
            float ov = __shfl_xor(minv[v], off, 32);
            int   oi = __shfl_xor(mini[v], off, 32);
            if (ov < minv[v]) { minv[v] = ov; mini[v] = oi; }
        }
    }
    if (lane == 0) {
        #pragma unroll
        for (int v = 0; v < 8; ++v) widx[w * 16 + v] = mini[v];       // rows 0..7
    }
    if (lane == 16) {
        #pragma unroll
        for (int v = 0; v < 8; ++v) widx[w * 16 + 8 + v] = mini[v];   // rows 8..15
    }
    __syncthreads();

    // --- one-hot scatter + histogram (buffer pre-zeroed by prep kernel) ---
    if (lane < 16) {
        int k = widx[w * 16 + lane];
        enc[(size_t)(n0 + lane) * K_CODES + k] = 1.0f;
        atomicAdd(&hist[k], 1u);
    }

    // --- gather quantized -> out (coalesced), accumulate commitment loss ---
    const int r    = lane & 15;                 // row of this lane
    const int kwin = widx[w * 16 + r];
    const int dh   = (lane >> 4) * 32;          // lanes 0-15: d 0..31, 16-31: 32..63
    float sq = 0.0f;
    #pragma unroll 4
    for (int j = 0; j < 32; ++j) {
        int d = dh + j;
        float q = embf[kwin * DIMV + d];
        size_t oidx = (size_t)bimg * CHW + (size_t)d * HWSZ + hw0 + r;
        float zv = z[oidx];
        outp[oidx] = q;                         // straight-through fwd == quantized
        float diff = q - zv;
        sq += diff * diff;
    }
    #pragma unroll
    for (int off = 16; off >= 1; off >>= 1) sq += __shfl_xor(sq, off, 32);
    if (lane == 0) atomicAdd(sqsum, sq);
}

// ---------------------------------------------------------------------------
// Kernel 3: finalize loss + perplexity scalars
// ---------------------------------------------------------------------------
__global__ void vq_finalize_kernel(const unsigned int* __restrict__ hist,
                                   const float* __restrict__ sqsum,
                                   float* __restrict__ outLP) {
    __shared__ float red[256];
    int t = threadIdx.x;
    float s = 0.0f;
    for (int k = t; k < K_CODES; k += 256) {
        float p = (float)hist[k] * (1.0f / (float)N_PTS);
        s += p * logf(p + 1e-10f);
    }
    red[t] = s;
    __syncthreads();
    for (int o = 128; o > 0; o >>= 1) {
        if (t < o) red[t] += red[t + o];
        __syncthreads();
    }
    if (t == 0) {
        outLP[0] = 0.25f * sqsum[0] * (1.0f / ((float)N_PTS * (float)DIMV)); // loss
        outLP[1] = expf(-red[0]);                                            // perplexity
    }
}

// ---------------------------------------------------------------------------
extern "C" void kernel_launch(void* const* d_in, const int* in_sizes, int n_in,
                              void* d_out, int out_size, void* d_ws, size_t ws_size,
                              hipStream_t stream) {
    (void)in_sizes; (void)n_in; (void)out_size; (void)ws_size;

    const float* z   = (const float*)d_in[0];   // [32,64,64,64]
    const float* emb = (const float*)d_in[1];   // [1024,64]

    float* outp  = (float*)d_out;               // [32,64,64,64]
    float* lossp = outp + OUT_ELEMS;            // loss, perplexity
    float* enc   = outp + OUT_ELEMS + 2;        // [131072,1024]

    // workspace carve-up
    _Float16*     emb_h = (_Float16*)d_ws;                        // 131072 B
    float*        e2    = (float*)((char*)d_ws + 131072);         // 4096 B
    unsigned int* hist  = (unsigned int*)((char*)d_ws + 135168);  // 4096 B
    float*        sqsum = (float*)((char*)d_ws + 139264);         // 4 B

    const size_t enc_pairs = (size_t)N_PTS * K_CODES / 2;         // float2 count

    vq_prep_kernel<<<2048, 256, 0, stream>>>(emb, (float2*)enc, enc_pairs,
                                             emb_h, e2, hist, sqsum);

    const size_t lds_bytes = (size_t)K_CODES * ROW_H * 2 + 4096 + 512; // 152064
    vq_main_kernel<<<1024, 256, lds_bytes, stream>>>(z, emb, emb_h, e2,
                                                     hist, sqsum, outp, enc);

    vq_finalize_kernel<<<1, 256, 0, stream>>>(hist, sqsum, lossp);
}
